// Attention_16776142258680
// MI455X (gfx1250) — matrix operand order repaired and checked by reading, not compiled, not run on hardware
//
#include <hip/hip_runtime.h>
#include <hip/hip_bf16.h>
#include <math.h>

typedef __attribute__((ext_vector_type(16))) _Float16 v16h;
typedef __attribute__((ext_vector_type(8)))  _Float16 v8h;
typedef __attribute__((ext_vector_type(8)))  float    v8f;

#define EMB     512
#define SEQ     4096
#define NB      2
#define NH      8
#define DH      64
#define NGROUPS 32
#define GEPS    100000.0f

// ---------------------------------------------------------------------------
// WMMA helpers (gfx1250 wave32, 16x16x32 f16 -> f32)
// ---------------------------------------------------------------------------
__device__ __forceinline__ v8f wmma16(v16h a, v16h b, v8f c) {
  // 8 args: (neg_a, A, neg_b, B, c_mod, C, reuse_a, reuse_b)
  return __builtin_amdgcn_wmma_f32_16x16x32_f16(false, a, false, b, (short)0, c,
                                                false, false);
}

// A fragment in this code's contiguous k order: lane row = lane&15; elements 0..15 = K[hf*16+i]
// (hf = lane>>4), matching the contiguous B operand loads
__device__ __forceinline__ v16h load_a_frag(const _Float16* row, int k0, int hf) {
  v8h lo = *(const v8h*)(row + k0 + hf * 16);
  v8h hi = *(const v8h*)(row + k0 + hf * 16 + 8);
  return __builtin_shufflevector(lo, hi, 0, 1, 2, 3, 4, 5, 6, 7,
                                 8, 9, 10, 11, 12, 13, 14, 15);
}

// ---------------------------------------------------------------------------
// GroupNorm stats: per (b,group) mean / rstd over contiguous 16ch x 4096.
// ---------------------------------------------------------------------------
__global__ void gn_stats_kernel(const float* __restrict__ in,
                                float* __restrict__ stats) {
  const int bg = blockIdx.x;  // b*NGROUPS + g
  const int N = (EMB / NGROUPS) * SEQ;
  const size_t base = (size_t)bg * N;
  float s = 0.f, s2 = 0.f;
  for (int i = threadIdx.x; i < N; i += 256) {
    float v = in[base + i];
    s += v;
    s2 += v * v;
  }
  __shared__ float rs[256], rq[256];
  rs[threadIdx.x] = s;
  rq[threadIdx.x] = s2;
  __syncthreads();
  for (int off = 128; off > 0; off >>= 1) {
    if ((int)threadIdx.x < off) {
      rs[threadIdx.x] += rs[threadIdx.x + off];
      rq[threadIdx.x] += rq[threadIdx.x + off];
    }
    __syncthreads();
  }
  if (threadIdx.x == 0) {
    float mu = rs[0] / (float)N;
    float var = rq[0] / (float)N - mu * mu;
    stats[bg * 2 + 0] = mu;
    stats[bg * 2 + 1] = rsqrtf(var + GEPS);
  }
}

// Normalize + affine + transpose (B,D,S) -> x16 (B,S,D) f16
__global__ void gn_apply_kernel(const float* __restrict__ in,
                                const float* __restrict__ stats,
                                const float* __restrict__ gamma,
                                const float* __restrict__ beta,
                                _Float16* __restrict__ x16) {
  size_t idx = (size_t)blockIdx.x * 256 + threadIdx.x;  // over B*D*S
  int b = (int)(idx / ((size_t)EMB * SEQ));
  int r = (int)(idx % ((size_t)EMB * SEQ));
  int d = r / SEQ;
  int si = r % SEQ;
  int g = d / (EMB / NGROUPS);
  float mu = stats[(b * NGROUPS + g) * 2 + 0];
  float rstd = stats[(b * NGROUPS + g) * 2 + 1];
  float v = (in[idx] - mu) * rstd * gamma[d] + beta[d];
  x16[((size_t)b * SEQ + si) * EMB + d] = (_Float16)v;
}

__global__ void f2h_kernel(const float* __restrict__ src,
                           _Float16* __restrict__ dst, int n) {
  int i = blockIdx.x * 256 + threadIdx.x;
  if (i < n) dst[i] = (_Float16)src[i];
}

// ---------------------------------------------------------------------------
// Projection GEMM:  acc[s][n] = x[b,s,:] . W[n,:] + bias[n], scaled, f16 store.
//   transposed==0 : out[B, NH, SEQ, DH]   (for Q, K)
//   transposed==1 : out[B, NH, DH, SEQ]   (for V -> contiguous-in-S B operand)
// Block = 8 waves, each wave owns 16 rows x 64 cols (4 WMMA C tiles).
// ---------------------------------------------------------------------------
__global__ void gemm_qkv_kernel(const _Float16* __restrict__ X,
                                const _Float16* __restrict__ Wh,
                                const float* __restrict__ bias,
                                _Float16* __restrict__ out, float scale,
                                int transposed) {
  const int lane = threadIdx.x & 31;
  const int wv_ = threadIdx.x >> 5;
  const int col = lane & 15;
  const int hf = lane >> 4;
  const int b = blockIdx.z;
  const int s0 = blockIdx.x * 128 + wv_ * 16;
  const int n0 = blockIdx.y * 64;

  const _Float16* xr = X + ((size_t)b * SEQ + s0 + col) * EMB;  // A row (M=lane&15)

  v8f c[4];
#pragma unroll
  for (int t = 0; t < 4; t++) {
    float bvv = bias[n0 + t * 16 + col];
#pragma unroll
    for (int j = 0; j < 8; j++) c[t][j] = bvv;
  }

#pragma unroll
  for (int k0 = 0; k0 < EMB; k0 += 32) {
    v16h a = load_a_frag(xr, k0, hf);
#pragma unroll
    for (int t = 0; t < 4; t++) {
      // B[k][n] = W[n][k]: row of W contiguous over k. b[i]=W[n][k0+hf*16+i]
      const _Float16* wr = Wh + (size_t)(n0 + t * 16 + col) * EMB + k0 + hf * 16;
      v16h bb = *(const v16h*)wr;
      c[t] = wmma16(a, bb, c[t]);
    }
  }

  if (!transposed) {
#pragma unroll
    for (int t = 0; t < 4; t++) {
      int n = n0 + t * 16 + col;
      int h = n >> 6, dh = n & 63;
#pragma unroll
      for (int j = 0; j < 8; j++) {
        int s = s0 + hf * 8 + j;  // C layout: lanes<16 rows 0..7, lanes>=16 rows 8..15
        out[(((size_t)(b * NH + h) * SEQ) + s) * DH + dh] =
            (_Float16)(c[t][j] * scale);
      }
    }
  } else {
    // C rows are contiguous in S -> packed 16B store per lane per tile
#pragma unroll
    for (int t = 0; t < 4; t++) {
      int n = n0 + t * 16 + col;
      int h = n >> 6, dh = n & 63;
      v8h pk;
#pragma unroll
      for (int j = 0; j < 8; j++) pk[j] = (_Float16)(c[t][j] * scale);
      *(v8h*)(out + ((size_t)(b * NH + h) * DH + dh) * SEQ + s0 + hf * 8) = pk;
    }
  }
}

// ---------------------------------------------------------------------------
// Flash attention per (b,h): wave = 16 queries, key blocks of 64
// (4 score tiles -> 16 WMMA per iteration; softmax fixed costs amortized).
// K  layout [B,NH,SEQ,DH]  -> QK^T B operand contiguous over d
// V^T layout [B,NH,DH,SEQ] -> P.V  B operand contiguous over key
// P round-trips through per-wave LDS; only wave-local DS ordering needed
// (s_wait_dscnt 0), so the loop has NO block barriers.
// ---------------------------------------------------------------------------
__global__ void attn_kernel(const _Float16* __restrict__ q16,
                            const _Float16* __restrict__ k16,
                            const _Float16* __restrict__ vT16,
                            _Float16* __restrict__ o16) {
  const int lane = threadIdx.x & 31;
  const int wv_ = threadIdx.x >> 5;
  const int col = lane & 15;
  const int hf = lane >> 4;
  const int b = blockIdx.z, h = blockIdx.y;
  const int s0 = blockIdx.x * 128 + wv_ * 16;
  const size_t headoff = (size_t)(b * NH + h) * SEQ * DH;  // == (b*NH+h)*DH*SEQ

  __shared__ __align__(16) _Float16 Pt[8][16 * 64];  // per-wave P staging (16x64)

  // Q fragments (q already scaled by 1/sqrt(dh) at projection)
  const _Float16* qr = q16 + headoff + (size_t)(s0 + col) * DH;
  v16h qa0 = load_a_frag(qr, 0, hf);
  v16h qa1 = load_a_frag(qr, 32, hf);

  const _Float16* kbase = k16 + headoff;   // [s][DH]
  const _Float16* vbase = vT16 + headoff;  // [DH][SEQ]

  v8f oc[4];
#pragma unroll
  for (int t = 0; t < 4; t++) oc[t] = (v8f){};
  float mrow[8], lrow[8];
#pragma unroll
  for (int j = 0; j < 8; j++) {
    mrow[j] = -3.0e38f;
    lrow[j] = 0.f;
  }

  for (int kb = 0; kb < SEQ; kb += 64) {
    // ---- scores: four 16x16 key tiles, 2 WMMA each (d = 0..31, 32..63) ----
    v8f sc[4];
#pragma unroll
    for (int t = 0; t < 4; t++) {
      const _Float16* kr = kbase + (size_t)(kb + t * 16 + col) * DH;
      if (kb + 64 < SEQ) __builtin_prefetch(kr + 64 * DH, 0, 1);  // next block
      v8f s = (v8f){};
      s = wmma16(qa0, *(const v16h*)(kr + hf * 16), s);
      s = wmma16(qa1, *(const v16h*)(kr + 32 + hf * 16), s);
      sc[t] = s;
    }

    // ---- online softmax: row M lives across one 16-lane group per C VGPR --
#pragma unroll
    for (int j = 0; j < 8; j++) {
      float mx = fmaxf(fmaxf(sc[0][j], sc[1][j]), fmaxf(sc[2][j], sc[3][j]));
#pragma unroll
      for (int off = 1; off < 16; off <<= 1)
        mx = fmaxf(mx, __shfl_xor(mx, off, 32));
      float mnew = fmaxf(mrow[j], mx);
      float corr = __expf(mrow[j] - mnew);
      float e0 = __expf(sc[0][j] - mnew);
      float e1 = __expf(sc[1][j] - mnew);
      float e2 = __expf(sc[2][j] - mnew);
      float e3 = __expf(sc[3][j] - mnew);
      float rsum = (e0 + e1) + (e2 + e3);
#pragma unroll
      for (int off = 1; off < 16; off <<= 1) rsum += __shfl_xor(rsum, off, 32);
      lrow[j] = lrow[j] * corr + rsum;
      mrow[j] = mnew;
      sc[0][j] = e0;  // reuse score regs as P
      sc[1][j] = e1;
      sc[2][j] = e2;
      sc[3][j] = e3;
#pragma unroll
      for (int t = 0; t < 4; t++) oc[t][j] *= corr;
    }

    // ---- C layout -> LDS row-major P (16x64 f16), wave-local region -------
    {
      _Float16* pw = &Pt[wv_][0];
#pragma unroll
      for (int j = 0; j < 8; j++) {
        int rbase = (hf * 8 + j) * 64;
#pragma unroll
        for (int t = 0; t < 4; t++)
          pw[rbase + t * 16 + col] = (_Float16)sc[t][j];
      }
    }
    // wave-internal cross-lane visibility: our own DS stores must land
    asm volatile("s_wait_dscnt 0" ::: "memory");

    // ---- O += P (16x64, two A-frags from LDS) x V^T (global, contiguous) --
    {
      const _Float16* pw = &Pt[wv_][0] + col * 64;
      v16h pa0 = load_a_frag(pw, 0, hf);   // keys kb+0..31
      v16h pa1 = load_a_frag(pw, 32, hf);  // keys kb+32..63
#pragma unroll
      for (int t = 0; t < 4; t++) {
        const _Float16* vr = vbase + (size_t)(t * 16 + col) * SEQ + kb + hf * 16;
        oc[t] = wmma16(pa0, *(const v16h*)vr, oc[t]);
        oc[t] = wmma16(pa1, *(const v16h*)(vr + 32), oc[t]);
      }
    }
  }

  // normalize and store into o16 [B, S, D] (D = h*64 + dh) for output GEMM
#pragma unroll
  for (int t = 0; t < 4; t++) {
    int dh = t * 16 + col;
#pragma unroll
    for (int j = 0; j < 8; j++) {
      int s = s0 + hf * 8 + j;
      float val = oc[t][j] / lrow[j];
      o16[((size_t)b * SEQ + s) * EMB + h * DH + dh] = (_Float16)val;
    }
  }
}

// ---------------------------------------------------------------------------
// Output projection + residual, transposed store to (B, D, S).
// C-layout rows are contiguous in S -> 32B vector add/store per lane/tile.
// ---------------------------------------------------------------------------
__global__ void gemm_out_kernel(const _Float16* __restrict__ X,
                                const _Float16* __restrict__ Wh,
                                const float* __restrict__ bias,
                                const float* __restrict__ resid,
                                float* __restrict__ out) {
  const int lane = threadIdx.x & 31;
  const int wv_ = threadIdx.x >> 5;
  const int col = lane & 15;
  const int hf = lane >> 4;
  const int b = blockIdx.z;
  const int s0 = blockIdx.x * 128 + wv_ * 16;
  const int n0 = blockIdx.y * 64;

  const _Float16* xr = X + ((size_t)b * SEQ + s0 + col) * EMB;

  v8f c[4];
#pragma unroll
  for (int t = 0; t < 4; t++) {
    float bvv = bias[n0 + t * 16 + col];
#pragma unroll
    for (int j = 0; j < 8; j++) c[t][j] = bvv;
  }

#pragma unroll
  for (int k0 = 0; k0 < EMB; k0 += 32) {
    v16h a = load_a_frag(xr, k0, hf);
#pragma unroll
    for (int t = 0; t < 4; t++) {
      const _Float16* wr = Wh + (size_t)(n0 + t * 16 + col) * EMB + k0 + hf * 16;
      v16h bb = *(const v16h*)wr;
      c[t] = wmma16(a, bb, c[t]);
    }
  }

#pragma unroll
  for (int t = 0; t < 4; t++) {
    int d = n0 + t * 16 + col;
    size_t base = ((size_t)b * EMB + d) * SEQ + s0 + hf * 8;
    v8f r = *(const v8f*)(resid + base);
    *(v8f*)(out + base) = c[t] + r;
  }
}

// ---------------------------------------------------------------------------
extern "C" void kernel_launch(void* const* d_in, const int* in_sizes, int n_in,
                              void* d_out, int out_size, void* d_ws,
                              size_t ws_size, hipStream_t stream) {
  (void)in_sizes; (void)n_in; (void)out_size; (void)ws_size;
  const float* input = (const float*)d_in[0];
  const float* gamma = (const float*)d_in[1];
  const float* beta = (const float*)d_in[2];
  const float* wq = (const float*)d_in[3];
  const float* bq = (const float*)d_in[4];
  const float* wk = (const float*)d_in[5];
  const float* bk = (const float*)d_in[6];
  const float* wv = (const float*)d_in[7];
  const float* bvp = (const float*)d_in[8];
  const float* wo = (const float*)d_in[9];
  const float* bo = (const float*)d_in[10];
  float* out = (float*)d_out;

  char* ws = (char*)d_ws;
  size_t off = 0;
  float* stats = (float*)(ws + off); off += 1024;
  const size_t HB = (size_t)NB * SEQ * EMB * sizeof(_Float16);  // 8 MB
  _Float16* x16 = (_Float16*)(ws + off); off += HB;
  _Float16* q16 = (_Float16*)(ws + off); off += HB;
  _Float16* k16 = (_Float16*)(ws + off); off += HB;
  _Float16* vT16 = (_Float16*)(ws + off); off += HB;
  _Float16* o16 = (_Float16*)(ws + off); off += HB;
  const size_t WB = (size_t)EMB * EMB * sizeof(_Float16);       // 512 KB
  _Float16* wq16 = (_Float16*)(ws + off); off += WB;
  _Float16* wk16 = (_Float16*)(ws + off); off += WB;
  _Float16* wv16 = (_Float16*)(ws + off); off += WB;
  _Float16* wo16 = (_Float16*)(ws + off); off += WB;

  gn_stats_kernel<<<NB * NGROUPS, 256, 0, stream>>>(input, stats);
  gn_apply_kernel<<<(NB * EMB * SEQ) / 256, 256, 0, stream>>>(input, stats,
                                                              gamma, beta, x16);
  f2h_kernel<<<(EMB * EMB) / 256, 256, 0, stream>>>(wq, wq16, EMB * EMB);
  f2h_kernel<<<(EMB * EMB) / 256, 256, 0, stream>>>(wk, wk16, EMB * EMB);
  f2h_kernel<<<(EMB * EMB) / 256, 256, 0, stream>>>(wv, wv16, EMB * EMB);
  f2h_kernel<<<(EMB * EMB) / 256, 256, 0, stream>>>(wo, wo16, EMB * EMB);

  dim3 gg(SEQ / 128, EMB / 64, NB);
  gemm_qkv_kernel<<<gg, 256, 0, stream>>>(x16, wq16, bq, q16, 0.125f, 0);
  gemm_qkv_kernel<<<gg, 256, 0, stream>>>(x16, wk16, bk, k16, 1.0f, 0);
  gemm_qkv_kernel<<<gg, 256, 0, stream>>>(x16, wv16, bvp, vT16, 1.0f, 1);

  dim3 ga(SEQ / 128, NH, NB);
  attn_kernel<<<ga, 256, 0, stream>>>(q16, k16, vT16, o16);

  gemm_out_kernel<<<gg, 256, 0, stream>>>(o16, wo16, bo, input, out);
}
